// ParallelGraphSCAttention_3977139716841
// MI455X (gfx1250) — matrix operand order repaired and checked by reading, not compiled, not run on hardware
//
#include <hip/hip_runtime.h>

#define N_ROWS 262144
#define B_SEG  1024
#define SPLIT  4      // main_pass blocks per segment
#define GSPLIT 4      // gap_stats blocks per segment
#define PD     4      // software pipeline depth (rows in flight per block)
#define EPS    1e-5f

typedef __attribute__((ext_vector_type(16))) _Float16 v16h;
typedef __attribute__((ext_vector_type(8)))  float    v8f;

// ---------------- kernel 1: segment boundaries (batch is sorted) -------------
__global__ void seg_bounds(const int* __restrict__ batch, int* __restrict__ seg) {
    int b = blockIdx.x * blockDim.x + threadIdx.x;
    if (b > B_SEG) return;
    int lo = 0, hi = N_ROWS;
    while (lo < hi) { int mid = (lo + hi) >> 1; if (batch[mid] < b) lo = mid + 1; else hi = mid; }
    seg[b] = lo;
}

// ---------------- kernel 2: per-segment mean/max of x_c (split partials) -----
__global__ __launch_bounds__(256)
void gap_stats(const float* __restrict__ x, const int* __restrict__ seg,
               float* __restrict__ psum, float* __restrict__ pmax) {
    const int bid = blockIdx.x;
    const int b = bid / GSPLIT, q = bid % GSPLIT;
    const int t = threadIdx.x;
    const int g = t >> 4, h = t & 15;
    const int s0 = seg[b], s1 = seg[b + 1], len = s1 - s0;
    const int r0 = s0 + (len * q) / GSPLIT;
    const int r1 = s0 + (len * (q + 1)) / GSPLIT;
    const size_t coff = (size_t)(g * 32 + h);

    float sum = 0.f, mx = -3.4e38f;
    int r = r0;
    for (; r + 4 <= r1; r += 4) {            // 4 independent loads in flight
        float v0 = x[(size_t)(r + 0) * 512 + coff];
        float v1 = x[(size_t)(r + 1) * 512 + coff];
        float v2 = x[(size_t)(r + 2) * 512 + coff];
        float v3 = x[(size_t)(r + 3) * 512 + coff];
        sum += (v0 + v1) + (v2 + v3);
        mx = fmaxf(fmaxf(fmaxf(mx, v0), fmaxf(v1, v2)), v3);
    }
    for (; r < r1; ++r) {
        float v = x[(size_t)r * 512 + coff];
        sum += v; mx = fmaxf(mx, v);
    }
    psum[(size_t)bid * 256 + t] = sum;
    pmax[(size_t)bid * 256 + t] = mx;
}

// ---------------- kernel 2b: fold partials into gap features -----------------
__global__ void gap_combine(const int* __restrict__ seg,
                            const float* __restrict__ psum, const float* __restrict__ pmax,
                            float* __restrict__ gap) {
    int idx = blockIdx.x * blockDim.x + threadIdx.x;
    if (idx >= B_SEG * 256) return;
    int b = idx >> 8, t = idx & 255;
    float sum = 0.f, mx = -3.4e38f;
    for (int q = 0; q < GSPLIT; ++q) {
        sum += psum[(size_t)(b * GSPLIT + q) * 256 + t];
        mx = fmaxf(mx, pmax[(size_t)(b * GSPLIT + q) * 256 + t]);
    }
    float cnt = (float)(seg[b + 1] - seg[b]);
    gap[(size_t)b * 512 + t]       = sum / fmaxf(cnt, 1.0f);
    gap[(size_t)b * 512 + 256 + t] = (cnt > 0.f) ? mx : 0.f;
}

// ---------------- kernel 3: channel MLP via WMMA (wave32) --------------------
// GEMM1: (16 x 512) @ (512 x 64) + bias, ReLU;  GEMM2: (16 x 64) @ (64 x 256)
__global__ __launch_bounds__(32)
void ch_mlp_wmma(const float* __restrict__ gap,
                 const float* __restrict__ W1, const float* __restrict__ b1,
                 const float* __restrict__ W2, const float* __restrict__ b2,
                 const float* __restrict__ cweight, const float* __restrict__ cbias,
                 float* __restrict__ gates) {
    __shared__ float hch_s[16 * 64];
    const int tile = blockIdx.x;
    const int lane = threadIdx.x;
    const int half = lane >> 4;
    const int l16  = lane & 15;
    const int rowA = tile * 16 + l16;

    v8f acc[4];
    #pragma unroll
    for (int nt = 0; nt < 4; ++nt)
        #pragma unroll
        for (int i = 0; i < 8; ++i) acc[nt][i] = 0.f;

    for (int kt = 0; kt < 16; ++kt) {             // K = 512 in chunks of 32
        v16h a;
        #pragma unroll
        for (int e = 0; e < 16; ++e) {            // A layout: lane<16 K={0..7,16..23}
            int k = kt * 32 + (e & 7) + ((e >> 3) << 4) + half * 8;
            a[e] = (_Float16)gap[(size_t)rowA * 512 + k];
        }
        #pragma unroll
        for (int nt = 0; nt < 4; ++nt) {
            v16h bm;
            #pragma unroll
            for (int e = 0; e < 16; ++e) {        // B layout: lane = ((k/8)%2)*16 + n
                int k = kt * 32 + (e & 7) + ((e >> 3) << 4) + half * 8;
                bm[e] = (_Float16)W1[(size_t)k * 64 + nt * 16 + l16];
            }
            acc[nt] = __builtin_amdgcn_wmma_f32_16x16x32_f16(
                false, a, false, bm, (short)0, acc[nt], false, false);
        }
    }
    #pragma unroll
    for (int nt = 0; nt < 4; ++nt)
        #pragma unroll
        for (int rr = 0; rr < 8; ++rr) {          // C layout: M = rr + half*8
            int M = rr + half * 8, nc = nt * 16 + l16;
            hch_s[M * 64 + nc] = fmaxf(acc[nt][rr] + b1[nc], 0.f);
        }
    __syncthreads();

    v16h a2[2];
    #pragma unroll
    for (int kt = 0; kt < 2; ++kt)
        #pragma unroll
        for (int e = 0; e < 16; ++e) {
            int k = kt * 32 + (e & 7) + ((e >> 3) << 4) + half * 8;
            a2[kt][e] = (_Float16)hch_s[l16 * 64 + k];
        }

    for (int nt = 0; nt < 16; ++nt) {             // N = 256 in tiles of 16
        v8f acc2;
        #pragma unroll
        for (int i = 0; i < 8; ++i) acc2[i] = 0.f;
        #pragma unroll
        for (int kt = 0; kt < 2; ++kt) {
            v16h bm;
            #pragma unroll
            for (int e = 0; e < 16; ++e) {
                int k = kt * 32 + (e & 7) + ((e >> 3) << 4) + half * 8;
                bm[e] = (_Float16)W2[(size_t)k * 256 + nt * 16 + l16];
            }
            acc2 = __builtin_amdgcn_wmma_f32_16x16x32_f16(
                false, a2[kt], false, bm, (short)0, acc2, false, false);
        }
        int nc = nt * 16 + l16;
        float cw = 1.0f + cweight[nc & 15];
        float cb = cbias[nc & 15];
        #pragma unroll
        for (int rr = 0; rr < 8; ++rr) {
            int M = rr + half * 8;
            float v  = acc2[rr] + b2[nc];
            float sg = 1.0f / (1.0f + __expf(-v));
            gates[(size_t)(tile * 16 + M) * 256 + nc] = sg * cw + cb;
        }
    }
}

// ---------------- kernels 4/6: main row pass (templated) ---------------------
// PHASE 0: accumulate segment sums of out, out^2.  PHASE 1: apply affine + permuted store.
template <int PHASE>
__global__ __launch_bounds__(256)
void main_pass(const float* __restrict__ x, const int* __restrict__ seg,
               const float* __restrict__ gates,
               const float* __restrict__ gn_w, const float* __restrict__ gn_b,
               const float* __restrict__ spW1, const float* __restrict__ spb1,
               const float* __restrict__ spW2, const float* __restrict__ spb2,
               const float* __restrict__ sweight, const float* __restrict__ sbias,
               const float* __restrict__ init_scale,
               const float* __restrict__ coefA, const float* __restrict__ coefB,
               float* __restrict__ s1p, float* __restrict__ s2p,
               float* __restrict__ out) {
    __shared__ float gate_s[256];
    __shared__ float xn_s[256];
    __shared__ float sc_s[48];
    __shared__ float hid_s[8];
    __shared__ float out_s[512];

    const int bid = blockIdx.x;
    const int b = bid / SPLIT, q = bid % SPLIT;
    const int t = threadIdx.x;
    const int g = t >> 4, h = t & 15;
    const int s0 = seg[b], s1 = seg[b + 1];
    const int len = s1 - s0;
    const int r0 = s0 + (len * q) / SPLIT;
    const int r1 = s0 + (len * (q + 1)) / SPLIT;

    gate_s[t] = gates[(size_t)b * 256 + t];
    const int   c1  = g * 32 + h, c2 = c1 + 16;     // x_c / x_s channels owned
    const float gnw = gn_w[t], gnb = gn_b[t];
    const float swh = 1.0f + sweight[h], sbh = sbias[h];
    const float ts  = tanhf(init_scale[0]);
    float cA1 = 0.f, cB1 = 0.f, cA2 = 0.f, cB2 = 0.f;
    if (PHASE == 1) {
        cA1 = coefA[(size_t)b * 512 + c1]; cB1 = coefB[(size_t)b * 512 + c1];
        cA2 = coefA[(size_t)b * 512 + c2]; cB2 = coefB[(size_t)b * 512 + c2];
    }
    float a1 = 0.f, a1s = 0.f, a2 = 0.f, a2s = 0.f;

    // -------- software pipeline: PD rows of loads in flight ------------------
    float vcb[PD], vsb[PD];
    #pragma unroll
    for (int i = 0; i < PD; ++i) {
        const int rr = r0 + i;
        vcb[i] = (rr < r1) ? x[(size_t)rr * 512 + c1] : 0.f;
        vsb[i] = (rr < r1) ? x[(size_t)rr * 512 + c2] : 0.f;
    }
    __syncthreads();

    for (int rbase = r0; rbase < r1; rbase += PD) {
        #pragma unroll
        for (int i = 0; i < PD; ++i) {
            const int r = rbase + i;
            if (r >= r1) break;                       // uniform across block
            const float vc = vcb[i];
            const float vs = vsb[i];
            const int rn = r + PD;                    // refill slot i
            if (rn < r1) {
                vcb[i] = x[(size_t)rn * 512 + c1];
                vsb[i] = x[(size_t)rn * 512 + c2];
            }
            if (i == 0 && rbase + 2 * PD < r1)        // warm L2 ahead of pipeline
                __builtin_prefetch(&x[(size_t)(rbase + 2 * PD) * 512 + c1], 0, 0);

            // group-norm over H=16 (16-lane cluster, wave32 shuffles)
            float sA = vs, sB = vs * vs;
            #pragma unroll
            for (int m = 1; m < 16; m <<= 1) {
                sA += __shfl_xor(sA, m, 32);
                sB += __shfl_xor(sB, m, 32);
            }
            const float mu  = sA * (1.0f / 16.0f);
            const float var = sB * (1.0f / 16.0f) - mu * mu;
            const float xn  = (vs - mu) * rsqrtf(var + EPS) * gnw + gnb;

            __syncthreads();                 // protect xn_s/sc_s/out_s reuse
            xn_s[t] = xn;
            __syncthreads();

            // cross-group stats for this h
            float m0 = 0.f, mx = -3.4e38f;
            #pragma unroll
            for (int gg = 0; gg < 16; ++gg) {
                float v = xn_s[gg * 16 + h];
                m0 += v; mx = fmaxf(mx, v);
            }
            m0 *= (1.0f / 16.0f);
            float ssq = 0.f;
            #pragma unroll
            for (int gg = 0; gg < 16; ++gg) {
                float v = xn_s[gg * 16 + h] - m0;
                ssq += v * v;
            }
            const float sd = sqrtf(ssq * (1.0f / 15.0f));
            if (g == 0) { sc_s[h] = m0; sc_s[16 + h] = mx; sc_s[32 + h] = sd; }
            __syncthreads();

            if (t < 8) {                     // spatial MLP hidden (48 -> 8)
                float acc = spb1[t];
                #pragma unroll
                for (int ii = 0; ii < 48; ++ii) acc += sc_s[ii] * spW1[ii * 8 + t];
                hid_s[t] = fmaxf(acc, 0.f);
            }
            __syncthreads();

            float ga = spb2[h];              // spatial gate (8 -> 16)
            #pragma unroll
            for (int j = 0; j < 8; ++j) ga += hid_s[j] * spW2[j * 16 + h];
            const float spg = (1.0f / (1.0f + __expf(-ga))) * swh + sbh;

            const float o1 = vc * gate_s[t] + ts * vc;
            const float o2 = vs * spg       + ts * vs;

            if (PHASE == 0) {
                a1 += o1; a1s += o1 * o1; a2 += o2; a2s += o2 * o2;
            } else {
                const float f1 = cA1 * o1 + cB1;
                const float f2 = cA2 * o2 + cB2;
                out_s[2 * (c1 & 255) + (c1 >> 8)] = f1;   // interleave permutation
                out_s[2 * (c2 & 255) + (c2 >> 8)] = f2;
                __syncthreads();
                ((float2*)(out + (size_t)r * 512))[t] = ((const float2*)out_s)[t];
            }
        }
    }
    if (PHASE == 0) {
        s1p[(size_t)bid * 512 + c1] = a1;  s2p[(size_t)bid * 512 + c1] = a1s;
        s1p[(size_t)bid * 512 + c2] = a2;  s2p[(size_t)bid * 512 + c2] = a2s;
    }
}

// ---------------- kernel 5: fold sums into per-(segment,channel) affine ------
__global__ void coef_kernel(const int* __restrict__ seg,
                            const float* __restrict__ s1p, const float* __restrict__ s2p,
                            const float* __restrict__ gms,
                            const float* __restrict__ gnw, const float* __restrict__ gnb,
                            float* __restrict__ coefA, float* __restrict__ coefB) {
    int idx = blockIdx.x * blockDim.x + threadIdx.x;
    if (idx >= B_SEG * 512) return;
    int b = idx >> 9, c = idx & 511;
    float cnt = (float)(seg[b + 1] - seg[b]);
    float inv = 1.0f / fmaxf(cnt, 1.0f);
    float S1 = 0.f, S2 = 0.f;
    for (int qq = 0; qq < SPLIT; ++qq) {
        S1 += s1p[(size_t)(b * SPLIT + qq) * 512 + c];
        S2 += s2p[(size_t)(b * SPLIT + qq) * 512 + c];
    }
    float mean = S1 * inv;
    float m    = mean * gms[c];
    float var  = (S2 - 2.f * m * S1 + cnt * m * m) * inv;
    float A    = gnw[c] * rsqrtf(var + EPS);
    coefA[idx] = A;
    coefB[idx] = gnb[c] - A * m;
}

// ---------------- host launcher ---------------------------------------------
extern "C" void kernel_launch(void* const* d_in, const int* in_sizes, int n_in,
                              void* d_out, int out_size, void* d_ws, size_t ws_size,
                              hipStream_t stream) {
    const float* x       = (const float*)d_in[0];
    const int*   batch   = (const int*)d_in[1];
    // d_in[2] = num_segments (fixed B=1024)
    const float* cweight = (const float*)d_in[3];
    const float* cbias   = (const float*)d_in[4];
    const float* sweight = (const float*)d_in[5];
    const float* sbias   = (const float*)d_in[6];
    const float* ch_W1   = (const float*)d_in[7];
    const float* ch_b1   = (const float*)d_in[8];
    const float* ch_W2   = (const float*)d_in[9];
    const float* ch_b2   = (const float*)d_in[10];
    const float* gn_w    = (const float*)d_in[11];
    const float* gn_b    = (const float*)d_in[12];
    const float* spW1    = (const float*)d_in[13];
    const float* spb1    = (const float*)d_in[14];
    const float* spW2    = (const float*)d_in[15];
    const float* spb2    = (const float*)d_in[16];
    const float* init_sc = (const float*)d_in[17];
    const float* gnorm_w = (const float*)d_in[18];
    const float* gnorm_b = (const float*)d_in[19];
    const float* gnorm_ms= (const float*)d_in[20];
    float* out = (float*)d_out;

    // Workspace layout (~23 MB). s1p aliases the dead gap-partials region:
    //   [seg 8KB][P: psum 4MB | pmax 4MB  -> later s1p 8MB][gap 2MB][gates 1MB]
    //   [s2p 8MB][cA 2MB][cB 2MB]
    char* ws = (char*)d_ws;
    int*   seg   = (int*)ws;
    float* psum  = (float*)(ws + 8192);
    float* pmax  = (float*)((char*)psum  + (size_t)B_SEG * GSPLIT * 256 * 4);
    float* s1p   = psum;  // alias: gap partials are consumed before phase-0 runs
    float* gap   = (float*)((char*)pmax  + (size_t)B_SEG * GSPLIT * 256 * 4);
    float* gates = (float*)((char*)gap   + (size_t)B_SEG * 512 * 4);
    float* s2p   = (float*)((char*)gates + (size_t)B_SEG * 256 * 4);
    float* cA    = (float*)((char*)s2p   + (size_t)B_SEG * SPLIT * 512 * 4);
    float* cB    = (float*)((char*)cA    + (size_t)B_SEG * 512 * 4);

    seg_bounds<<<dim3((B_SEG + 64) / 64), dim3(64), 0, stream>>>(batch, seg);
    gap_stats<<<dim3(B_SEG * GSPLIT), dim3(256), 0, stream>>>(x, seg, psum, pmax);
    gap_combine<<<dim3((B_SEG * 256) / 256), dim3(256), 0, stream>>>(seg, psum, pmax, gap);
    ch_mlp_wmma<<<dim3(B_SEG / 16), dim3(32), 0, stream>>>(
        gap, ch_W1, ch_b1, ch_W2, ch_b2, cweight, cbias, gates);
    main_pass<0><<<dim3(B_SEG * SPLIT), dim3(256), 0, stream>>>(
        x, seg, gates, gn_w, gn_b, spW1, spb1, spW2, spb2, sweight, sbias,
        init_sc, (const float*)nullptr, (const float*)nullptr, s1p, s2p, (float*)nullptr);
    coef_kernel<<<dim3((B_SEG * 512) / 256), dim3(256), 0, stream>>>(
        seg, s1p, s2p, gnorm_ms, gnorm_w, gnorm_b, cA, cB);
    main_pass<1><<<dim3(B_SEG * SPLIT), dim3(256), 0, stream>>>(
        x, seg, gates, gn_w, gn_b, spW1, spb1, spW2, spb2, sweight, sbias,
        init_sc, cA, cB, (float*)nullptr, (float*)nullptr, out);
}
